// FactorizedAttentionBlock_19679540150350
// MI455X (gfx1250) — matrix operand order repaired and checked by reading, not compiled, not run on hardware
//
#include <hip/hip_runtime.h>
#include <hip/hip_bf16.h>
#include <math.h>

// ---------------------------------------------------------------------------
// FactorizedAttentionBlock for MI455X (gfx1250, wave32, WMMA bf16)
//   x:[8,256,64,64] f32 ; WS=8, NH=8, hd=32 ; tokens M = 512*64 = 32768
// All GEMMs + attention run on v_wmma_f32_16x16x32_bf16 (f32 accumulate).
// ---------------------------------------------------------------------------

typedef __attribute__((ext_vector_type(16))) __bf16 v16bf;
typedef __attribute__((ext_vector_type(8)))  __bf16 v8bf;
typedef __attribute__((ext_vector_type(8)))  float  v8f;

#define B_    8
#define C_    256
#define H_    64
#define W_    64
#define NH_   8
#define HD_   32
#define L_    512            // nwin*B
#define NTOK_ 64             // ws*ws
#define M_    (L_ * NTOK_)   // 32768 tokens
#define K3C_  768

__device__ __forceinline__ __bf16 to_bf16(float f) {
  unsigned u = __float_as_uint(f);
  u += 0x7FFFu + ((u >> 16) & 1u);               // round-to-nearest-even
  unsigned short s = (unsigned short)(u >> 16);
  return __builtin_bit_cast(__bf16, s);
}

__device__ __forceinline__ v8f wmma_bf16(v16bf a, v16bf b, v8f c) {
  return __builtin_amdgcn_wmma_f32_16x16x32_bf16(false, a, false, b,
                                                 (short)0, c, false, false);
}

// A fragment (16x32, row-major source, ld in elements). Contiguous k order used here (same as load_bT_frag):
// lane<16: M=lane, halves 0..15 = K 0..15
// lane>=16: M=lane-16, K offsets +16.
__device__ __forceinline__ v16bf load_a_frag(const __bf16* base, int ld) {
  int lane = threadIdx.x & 31;
  int row  = lane & 15;
  int kb   = (lane < 16) ? 0 : 16;
  const __bf16* p = base + (size_t)row * ld + kb;
  v8bf lo = *(const v8bf*)(p);
  v8bf hi = *(const v8bf*)(p + 8);
  return __builtin_shufflevector(lo, hi, 0,1,2,3,4,5,6,7,8,9,10,11,12,13,14,15);
}

// B fragment (32x16) where B[k][n] = Wt[n][k], Wt row-major [N][K] (k fastest).
// lane<16: N=lane, K=0..15 across halves; lane>=16: N=lane-16, K=16..31.
// -> one contiguous 32-byte load per lane.
__device__ __forceinline__ v16bf load_bT_frag(const __bf16* base, int ld) {
  int lane = threadIdx.x & 31;
  int n    = lane & 15;
  int kb   = (lane < 16) ? 0 : 16;
  return *(const v16bf*)(base + (size_t)n * ld + kb);
}

// --------------------------- SE gate --------------------------------------
__global__ __launch_bounds__(256) void pool_kernel(const float* __restrict__ x,
                                                   float* __restrict__ pooled) {
  __shared__ float red[256];
  int bc = blockIdx.x;                      // [0, B*C)
  int tid = threadIdx.x;
  const float* p = x + (size_t)bc * (H_ * W_);
  float s = 0.f;
  for (int i = tid; i < H_ * W_; i += 256) s += p[i];
  red[tid] = s;
  __syncthreads();
  for (int st = 128; st > 0; st >>= 1) {
    if (tid < st) red[tid] += red[tid + st];
    __syncthreads();
  }
  if (tid == 0) pooled[bc] = red[0] * (1.0f / (H_ * W_));
}

__global__ __launch_bounds__(256) void se_mlp_kernel(
    const float* __restrict__ pooled, const float* __restrict__ w1,
    const float* __restrict__ b1, const float* __restrict__ w2,
    const float* __restrict__ b2, float* __restrict__ ca) {
  __shared__ float h1[B_ * 64];
  int tid = threadIdx.x;
  for (int idx = tid; idx < B_ * 64; idx += 256) {
    int b = idx >> 6, j = idx & 63;
    float s = b1[j];
    for (int k = 0; k < C_; ++k) s += pooled[b * C_ + k] * w1[j * C_ + k];
    h1[idx] = fmaxf(s, 0.f);
  }
  __syncthreads();
  for (int idx = tid; idx < B_ * C_; idx += 256) {
    int b = idx >> 8, j = idx & 255;
    float s = b2[j];
    for (int k = 0; k < 64; ++k) s += h1[b * 64 + k] * w2[j * 64 + k];
    ca[idx] = 1.0f / (1.0f + __expf(-s));
  }
}

// ---- gate, window-partition permute, cast: A[m][c] = bf16(x*ca) -----------
__global__ __launch_bounds__(256) void gate_cast_kernel(
    const float* __restrict__ x, const float* __restrict__ ca,
    __bf16* __restrict__ A) {
  int i = blockIdx.x * 256 + threadIdx.x;   // ((b*C + c)*H + h)*W + w
  int w = i & 63, h = (i >> 6) & 63, c = (i >> 12) & 255, b = (i >> 20) & 7;
  int win = (h >> 3) * 8 + (w >> 3);
  int p   = (h & 7) * 8 + (w & 7);
  int m   = (win * B_ + b) * NTOK_ + p;
  A[(size_t)m * C_ + c] = to_bf16(x[i] * ca[b * C_ + c]);
}

__global__ __launch_bounds__(256) void cvt_bf16_kernel(const float* __restrict__ s,
                                                       __bf16* __restrict__ d, int n) {
  int i = blockIdx.x * 256 + threadIdx.x;
  if (i < n) d[i] = to_bf16(s[i]);
}

// ---- QKV GEMM: qkv[m][j] = A[m][:] . Win[j][:] + bin[j]  (M=32768,K=256,N=768)
// one wave -> 16(M) x 64(N) tile; 2048 M-tiles x 12 N-tiles = 24576 waves
__global__ __launch_bounds__(256) void qkv_gemm_kernel(
    const __bf16* __restrict__ A, const __bf16* __restrict__ Win,
    const float* __restrict__ bin, __bf16* __restrict__ qkv) {
  int wid  = blockIdx.x * 8 + (threadIdx.x >> 5);
  int mt   = wid % (M_ / 16);
  int nt   = wid / (M_ / 16);
  int m0   = mt * 16, n0 = nt * 64;
  int lane = threadIdx.x & 31;
  v8f acc[4] = {};
#pragma unroll
  for (int k0 = 0; k0 < C_; k0 += 32) {
    v16bf a = load_a_frag(A + (size_t)m0 * C_ + k0, C_);
#pragma unroll
    for (int j = 0; j < 4; ++j) {
      v16bf b = load_bT_frag(Win + (size_t)(n0 + 16 * j) * C_ + k0, C_);
      acc[j] = wmma_bf16(a, b, acc[j]);
    }
  }
  int half = lane >> 4, nn = lane & 15;
#pragma unroll
  for (int j = 0; j < 4; ++j) {
    int col = n0 + 16 * j + nn;
    float bias = bin[col];
#pragma unroll
    for (int r = 0; r < 8; ++r) {
      int row = m0 + r + half * 8;
      qkv[(size_t)row * K3C_ + col] = to_bf16(acc[j][r] + bias);
    }
  }
}

// ---- flash attention: one wave per (n, h, 16-row q tile) ------------------
// Q[l,d]=qkv[(l*64+n)*768 + h*32+d]; K at +256; V at +512. Token row stride 64*768.
__global__ __launch_bounds__(256) void attention_kernel(
    const __bf16* __restrict__ qkv, __bf16* __restrict__ obuf) {
  __shared__ __bf16 lds_p[8][16][32];   // P tile staging (D-layout -> A-layout)
  __shared__ __bf16 lds_v[8][32][32];   // V tile staging (row-major keys)

  int warp = threadIdx.x >> 5;
  int wid  = blockIdx.x * 8 + warp;
  int qt   = wid & 31;
  int h    = (wid >> 5) & 7;
  int n    = wid >> 8;                 // [0,64)
  int q0   = qt * 16;
  int lane = threadIdx.x & 31;
  int half = lane >> 4, nn = lane & 15;
  const int LDT = NTOK_ * K3C_;        // 49152: stride between consecutive l

  v16bf qa = load_a_frag(qkv + ((size_t)(q0 * NTOK_ + n)) * K3C_ + h * HD_, LDT);

  float mrow[8], lrow[8];
#pragma unroll
  for (int r = 0; r < 8; ++r) { mrow[r] = -INFINITY; lrow[r] = 0.f; }
  v8f o0 = {}, o1 = {};
  const v8f zero = {};
  const float scale = 0.17677669529663687f;   // 1/sqrt(32)

  for (int j0 = 0; j0 < L_; j0 += 32) {
    // scores for keys [j0, j0+32)
    const __bf16* kbase = qkv + ((size_t)(j0 * NTOK_ + n)) * K3C_ + C_ + h * HD_;
    v16bf kf0 = load_bT_frag(kbase, LDT);
    v16bf kf1 = load_bT_frag(kbase + (size_t)16 * LDT, LDT);
    v8f s0 = wmma_bf16(qa, kf0, zero);
    v8f s1 = wmma_bf16(qa, kf1, zero);

    // stage V tile [32 keys x 32 d] -> LDS, one row per lane
    {
      const __bf16* vrow =
          qkv + ((size_t)((j0 + lane) * NTOK_ + n)) * K3C_ + 2 * C_ + h * HD_;
      const v8bf* src = (const v8bf*)vrow;
      v8bf* dst = (v8bf*)&lds_v[warp][lane][0];
      dst[0] = src[0]; dst[1] = src[1]; dst[2] = src[2]; dst[3] = src[3];
    }

    // online softmax; rows live in 16-lane halves -> shfl_xor reductions
#pragma unroll
    for (int r = 0; r < 8; ++r) {
      float a0 = s0[r] * scale, a1 = s1[r] * scale;
      float t = fmaxf(a0, a1);
      t = fmaxf(t, __shfl_xor(t, 1, 32));
      t = fmaxf(t, __shfl_xor(t, 2, 32));
      t = fmaxf(t, __shfl_xor(t, 4, 32));
      t = fmaxf(t, __shfl_xor(t, 8, 32));
      float mnew = fmaxf(mrow[r], t);
      float corr = __expf(mrow[r] - mnew);
      float p0 = __expf(a0 - mnew), p1 = __expf(a1 - mnew);
      float ss = p0 + p1;
      ss += __shfl_xor(ss, 1, 32);
      ss += __shfl_xor(ss, 2, 32);
      ss += __shfl_xor(ss, 4, 32);
      ss += __shfl_xor(ss, 8, 32);
      lrow[r] = lrow[r] * corr + ss;
      mrow[r] = mnew;
      o0[r] *= corr; o1[r] *= corr;
      int rr = r + half * 8;
      lds_p[warp][rr][nn]      = to_bf16(p0);
      lds_p[warp][rr][nn + 16] = to_bf16(p1);
    }
    asm volatile("s_wait_dscnt 0" ::: "memory");   // wave-local LDS RAW

    // P as A-fragment (16 qrows x 32 keys)
    v16bf pa;
    {
      const __bf16* pp = &lds_p[warp][nn][half * 16];
      v8bf lo = *(const v8bf*)pp;
      v8bf hi = *(const v8bf*)(pp + 8);
      pa = __builtin_shufflevector(lo, hi, 0,1,2,3,4,5,6,7,8,9,10,11,12,13,14,15);
    }
    // V as B-fragments (32 keys x 16 d each)
    v16bf vb0, vb1;
    int kb3 = half * 16;
#pragma unroll
    for (int i = 0; i < 16; ++i) {
      vb0[i] = lds_v[warp][kb3 + i][nn];
      vb1[i] = lds_v[warp][kb3 + i][nn + 16];
    }
    o0 = wmma_bf16(pa, vb0, o0);
    o1 = wmma_bf16(pa, vb1, o1);
  }

  // normalize + store: obuf[m][h*32 + d]
#pragma unroll
  for (int r = 0; r < 8; ++r) {
    float inv = 1.0f / lrow[r];
    int qrow = q0 + r + half * 8;
    size_t m = (size_t)qrow * NTOK_ + n;
    obuf[m * C_ + h * HD_ + nn]      = to_bf16(o0[r] * inv);
    obuf[m * C_ + h * HD_ + 16 + nn] = to_bf16(o1[r] * inv);
  }
}

// ---- out projection + bias + residual (x*ca), write f32 output -----------
__global__ __launch_bounds__(256) void outproj_kernel(
    const __bf16* __restrict__ Ain, const __bf16* __restrict__ Wout,
    const float* __restrict__ bout, const float* __restrict__ x,
    const float* __restrict__ ca, float* __restrict__ out) {
  int wid  = blockIdx.x * 8 + (threadIdx.x >> 5);
  int mt   = wid % (M_ / 16);
  int nt   = wid / (M_ / 16);         // [0,4): 64-wide N tiles over C=256
  int m0   = mt * 16, n0 = nt * 64;
  int lane = threadIdx.x & 31;
  v8f acc[4] = {};
#pragma unroll
  for (int k0 = 0; k0 < C_; k0 += 32) {
    v16bf a = load_a_frag(Ain + (size_t)m0 * C_ + k0, C_);
#pragma unroll
    for (int j = 0; j < 4; ++j) {
      v16bf b = load_bT_frag(Wout + (size_t)(n0 + 16 * j) * C_ + k0, C_);
      acc[j] = wmma_bf16(a, b, acc[j]);
    }
  }
  int half = lane >> 4, nn = lane & 15;
#pragma unroll
  for (int j = 0; j < 4; ++j) {
    int col = n0 + 16 * j + nn;
    float bias = bout[col];
#pragma unroll
    for (int r = 0; r < 8; ++r) {
      int m = m0 + r + half * 8;
      int p = m & 63, l = m >> 6;
      int b = l & 7, win = l >> 3;
      int hh = ((win >> 3) << 3) + (p >> 3);
      int ww = ((win & 7) << 3) + (p & 7);
      size_t idx = (((size_t)b * C_ + col) * H_ + hh) * W_ + ww;
      out[idx] = x[idx] * ca[b * C_ + col] + acc[j][r] + bias;
    }
  }
}

// ---------------------------------------------------------------------------
extern "C" void kernel_launch(void* const* d_in, const int* in_sizes, int n_in,
                              void* d_out, int out_size, void* d_ws, size_t ws_size,
                              hipStream_t stream) {
  const float* x     = (const float*)d_in[0];
  const float* ca_w1 = (const float*)d_in[1];
  const float* ca_b1 = (const float*)d_in[2];
  const float* ca_w2 = (const float*)d_in[3];
  const float* ca_b2 = (const float*)d_in[4];
  const float* ipw   = (const float*)d_in[5];
  const float* ipb   = (const float*)d_in[6];
  const float* opw   = (const float*)d_in[7];
  const float* opb   = (const float*)d_in[8];
  float* out = (float*)d_out;

  char* ws = (char*)d_ws;
  size_t off = 0;
  float*  pooled = (float*)(ws + off);  off += (size_t)B_ * C_ * 4;        // 8 KB
  float*  ca     = (float*)(ws + off);  off += (size_t)B_ * C_ * 4;        // 8 KB
  __bf16* wq     = (__bf16*)(ws + off); off += (size_t)K3C_ * C_ * 2;      // 384 KB
  __bf16* wo     = (__bf16*)(ws + off); off += (size_t)C_ * C_ * 2;        // 128 KB
  __bf16* A      = (__bf16*)(ws + off); off += (size_t)M_ * C_ * 2;        // 16 MB
  __bf16* qkv    = (__bf16*)(ws + off); off += (size_t)M_ * K3C_ * 2;      // 48 MB
  __bf16* obuf   = (__bf16*)(ws + off); off += (size_t)M_ * C_ * 2;        // 16 MB

  pool_kernel<<<B_ * C_, 256, 0, stream>>>(x, pooled);
  se_mlp_kernel<<<1, 256, 0, stream>>>(pooled, ca_w1, ca_b1, ca_w2, ca_b2, ca);
  cvt_bf16_kernel<<<(K3C_ * C_ + 255) / 256, 256, 0, stream>>>(ipw, wq, K3C_ * C_);
  cvt_bf16_kernel<<<(C_ * C_ + 255) / 256, 256, 0, stream>>>(opw, wo, C_ * C_);
  gate_cast_kernel<<<(B_ * C_ * H_ * W_) / 256, 256, 0, stream>>>(x, ca, A);
  qkv_gemm_kernel<<<(M_ / 16) * 12 / 8, 256, 0, stream>>>(A, wq, ipb, qkv);
  attention_kernel<<<(NTOK_ * NH_ * (L_ / 16)) / 8, 256, 0, stream>>>(qkv, obuf);
  outproj_kernel<<<(M_ / 16) * 4 / 8, 256, 0, stream>>>(obuf, wo, opb, x, ca, out);
}
